// PatchSpatialAttentionModule_103079215470
// MI455X (gfx1250) — matrix operand order repaired and checked by reading, not compiled, not run on hardware
//
#include <hip/hip_runtime.h>

// ---------------------------------------------------------------------------
// PatchSpatialAttentionModule on MI455X (gfx1250)
// One workgroup (256 threads = 8 wave32) per patch (b, mh, mw).
// All GEMMs via v_wmma_f32_16x16x32_bf16 with fp32 accumulation.
// ---------------------------------------------------------------------------

#define NT 256
#define NW 8

typedef __bf16 bf16_t;
typedef __bf16 v16bf __attribute__((ext_vector_type(16)));
typedef float  v8f   __attribute__((ext_vector_type(8)));

static constexpr int Bb = 4, Cc = 256, Hh = 224, Ww = 224;
static constexpr int PH = 7, PW = 7;
static constexpr int Mh = 32, Mw = 32, Mm = Mh * Mw;   // 1024 patches / image
static constexpr int P  = 49, PP = 64;                 // tokens, padded tokens

// LDS row strides (padded to avoid bank conflicts; all even for dword loads)
static constexpr int ABS = 258;   // Ab  [64][258] bf16   (A, row-major p x c)
static constexpr int ATS = 66;    // Abt [256][66] bf16   (A^T, c x p)
static constexpr int SFS = 65;    // Sf  [64][65]  f32    (S -> Sc in place)
static constexpr int MCS = 258;   // Mcb [64][258] bf16   (Mc)
static constexpr int LBS = 66;    // Lb  [64][66]  bf16   (L)

struct LdsBlk {
  bf16_t Ab [PP * ABS];
  bf16_t Abt[Cc * ATS];
  float  Sf [PP * SFS];
  bf16_t Mcb[PP * MCS];
  bf16_t Lb [PP * LBS];
  float  meanA [Cc];
  float  meanMc[Cc];
};

// --- fragment loaders (layouts per CDNA5 ISA 7.12.2) -----------------------
// A-matrix 16x32 bf16: lane = hi*16+lo, m = lo, k = {hi*8 .. +8} U {16+hi*8 .. +8}
__device__ __forceinline__ v16bf afrag_bf(const bf16_t* base, int stride, int m, int kb) {
  union { v16bf v; unsigned u[8]; } r;
  const unsigned* row = (const unsigned*)(base + m * stride);
  int kw = kb >> 1;
#pragma unroll
  for (int e = 0; e < 4; ++e) r.u[e]     = row[kw + e];
#pragma unroll
  for (int e = 0; e < 4; ++e) r.u[4 + e] = row[kw + 8 + e];
  return r.v;
}

// B-matrix 32x16 bf16: lane n = lo, k = {hi*8 .. +8} U {16+hi*8 .. +8}, source row-contiguous
__device__ __forceinline__ v16bf bfrag_bf(const bf16_t* base, int stride, int n, int ks) {
  union { v16bf v; unsigned u[8]; } r;
  const unsigned* p = (const unsigned*)(base + n * stride + ks);
#pragma unroll
  for (int e = 0; e < 8; ++e) r.u[e] = p[(e & 3) + ((e >> 2) << 3)];
  return r.v;
}

// A-frag with on-the-fly f32 -> bf16 convert (for Sc)
__device__ __forceinline__ v16bf afrag_f32(const float* base, int stride, int m, int kb) {
  v16bf a;
  const float* row = base + m * stride;
#pragma unroll
  for (int e = 0; e < 8; ++e) a[e]     = (bf16_t)row[kb + e];
#pragma unroll
  for (int e = 0; e < 8; ++e) a[8 + e] = (bf16_t)row[kb + 16 + e];
  return a;
}

// A-frag mean-centered (Ac = A - meanA), bf16 source
__device__ __forceinline__ v16bf afrag_centered(const bf16_t* base, int stride,
                                                const float* mean, int m, int kb) {
  v16bf a;
  const bf16_t* row = base + m * stride;
#pragma unroll
  for (int e = 0; e < 8; ++e) { int k = kb + e;      a[e]     = (bf16_t)((float)row[k] - mean[k]); }
#pragma unroll
  for (int e = 0; e < 8; ++e) { int k = kb + 16 + e; a[8 + e] = (bf16_t)((float)row[k] - mean[k]); }
  return a;
}

// B-frag mean-centered (Mcc^T), bf16 source, row-contiguous in k
__device__ __forceinline__ v16bf bfrag_centered(const bf16_t* base, int stride,
                                                const float* mean, int n, int ks) {
  v16bf b;
  const bf16_t* row = base + n * stride;
#pragma unroll
  for (int e = 0; e < 16; ++e) { int k = ks + (e & 7) + ((e >> 3) << 4); b[e] = (bf16_t)((float)row[k] - mean[k]); }
  return b;
}

__device__ __forceinline__ v8f wmma_bf16(v16bf a, v16bf b, v8f c) {
  return __builtin_amdgcn_wmma_f32_16x16x32_bf16(false, a, false, b, (short)0, c, false, false);
}

__global__ __launch_bounds__(NT)
void psam_kernel(const float* __restrict__ x, const float* __restrict__ betap,
                 float* __restrict__ outp, float* __restrict__ scg,
                 float* __restrict__ covg, float* __restrict__ lg,
                 float* __restrict__ ecg) {
  __shared__ LdsBlk lds;

  const int tid  = threadIdx.x;
  const int lane = tid & 31;
  const int wid  = tid >> 5;
  const int lo   = lane & 15;
  const int hi   = lane >> 4;

  const int bid = blockIdx.x;            // = b*Mm + m
  const int b   = bid >> 10;
  const int m   = bid & (Mm - 1);
  const int mh  = m >> 5;
  const int mw  = m & 31;

  const float  beta   = betap[0];
  const size_t xbase  = ((size_t)b * Cc * Hh + (size_t)mh * PH) * Ww + (size_t)mw * PW;
  const size_t sqBase = (size_t)bid * (P * P);

  // ---- phase 0: zero pads, load A (bf16) into Ab and Abt --------------------
  for (int idx = tid; idx < 15 * (Cc / 2); idx += NT) {       // Ab rows 49..63
    int r = idx >> 7, cw = idx & 127;
    ((unsigned*)(lds.Ab + (P + r) * ABS))[cw] = 0u;
  }
  for (int idx = tid; idx < Cc * 15; idx += NT) {             // Abt cols 49..63
    int c = idx / 15, q = P + (idx - c * 15);
    lds.Abt[c * ATS + q] = (bf16_t)0.0f;
  }
  lds.meanMc[tid] = 0.0f;                                     // NT == Cc

  for (int e = tid; e < P * Cc; e += NT) {                    // coalesced-ish: j fastest
    int c  = e / P;
    int pe = e - c * P;
    int i  = pe / PW;
    int j  = pe - i * PW;
    float v = x[xbase + (size_t)c * (Hh * Ww) + i * Ww + j];
    bf16_t hv = (bf16_t)v;
    lds.Ab [pe * ABS + c] = hv;
    lds.Abt[c * ATS + pe] = hv;
  }
  __syncthreads();

  // ---- phase 1: S = A * A^T (64x64, K=256) ---------------------------------
  for (int t = wid; t < 16; t += NW) {
    int tm = t >> 2, tn = t & 3;
    v8f acc = {};
    for (int kk = 0; kk < Cc; kk += 32) {
      v16bf a  = afrag_bf(lds.Ab, ABS, tm * 16 + lo, kk + hi * 8);
      v16bf bb = bfrag_bf(lds.Ab, ABS, tn * 16 + lo, kk + hi * 8); // B[k][n]=A[n][k]
      acc = wmma_bf16(a, bb, acc);
    }
#pragma unroll
    for (int r = 0; r < 8; ++r)
      lds.Sf[(tm * 16 + hi * 8 + r) * SFS + tn * 16 + lo] = acc[r];
  }
  __syncthreads();

  // ---- phase 2: channel means of A; row softmax S -> Sc (fp32) -------------
  {
    float s = 0.0f;
    for (int p = 0; p < P; ++p) s += (float)lds.Ab[p * ABS + tid];
    lds.meanA[tid] = s * (1.0f / 49.0f);
  }
  if (tid < P) {
    float* row = lds.Sf + tid * SFS;
    float mx = row[0];
    for (int q = 1; q < P; ++q) mx = fmaxf(mx, row[q]);
    float s = 0.0f;
    for (int q = 0; q < P; ++q) { float e = __expf(row[q] - mx); row[q] = e; s += e; }
    float inv = 1.0f / s;
    for (int q = 0; q < P; ++q) row[q] *= inv;
    // cols 49..63 and rows 49..63 are already exactly 0 from the padded GEMM
  }
  __syncthreads();

  // ---- phase 3: write Sc; Mc = Sc * A (64x256, K=64); column sums ----------
  for (int e = tid; e < P * P; e += NT) {
    int p = e / P;
    scg[sqBase + e] = lds.Sf[p * SFS + (e - p * P)];
  }
  for (int t = wid; t < 64; t += NW) {
    int tm = t >> 4, tn = t & 15;
    v8f acc = {};
    for (int kk = 0; kk < PP; kk += 32) {
      v16bf a  = afrag_f32(lds.Sf, SFS, tm * 16 + lo, kk + hi * 8);
      v16bf bb = bfrag_bf (lds.Abt, ATS, tn * 16 + lo, kk + hi * 8); // B[q][c]=Abt[c][q]
      acc = wmma_bf16(a, bb, acc);
    }
    float csum = 0.0f;
#pragma unroll
    for (int r = 0; r < 8; ++r) {
      int p = tm * 16 + hi * 8 + r;
      lds.Mcb[p * MCS + tn * 16 + lo] = (bf16_t)acc[r];
      if (p < P) csum += acc[r];
    }
    atomicAdd(&lds.meanMc[tn * 16 + lo], csum);   // ds_add_f32
  }
  __syncthreads();
  lds.meanMc[tid] *= (1.0f / 49.0f);
  __syncthreads();

  // ---- phase 4: cov = Ac * Mcc^T / 49; L = Sc + cov ------------------------
  for (int t = wid; t < 16; t += NW) {
    int tm = t >> 2, tn = t & 3;
    v8f acc = {};
    for (int kk = 0; kk < Cc; kk += 32) {
      v16bf a  = afrag_centered(lds.Ab,  ABS, lds.meanA,  tm * 16 + lo, kk + hi * 8);
      v16bf bb = bfrag_centered(lds.Mcb, MCS, lds.meanMc, tn * 16 + lo, kk + hi * 8);
      acc = wmma_bf16(a, bb, acc);
    }
#pragma unroll
    for (int r = 0; r < 8; ++r) {
      int p = tm * 16 + hi * 8 + r;
      int q = tn * 16 + lo;
      float cv = acc[r] * (1.0f / 49.0f);
      float lv = cv + lds.Sf[p * SFS + q];
      lds.Lb[p * LBS + q] = (bf16_t)lv;
      if (p < P && q < P) {
        covg[sqBase + p * P + q] = cv;
        lg  [sqBase + p * P + q] = lv;
      }
    }
  }
  __syncthreads();

  // ---- phase 5: Ec = L * A (64x256, K=64); fused fold-back + out -----------
  for (int t = wid; t < 64; t += NW) {
    int tm = t >> 4, tn = t & 15;
    v8f acc = {};
    for (int kk = 0; kk < PP; kk += 32) {
      v16bf a  = afrag_bf(lds.Lb,  LBS, tm * 16 + lo, kk + hi * 8);
      v16bf bb = bfrag_bf(lds.Abt, ATS, tn * 16 + lo, kk + hi * 8);
      acc = wmma_bf16(a, bb, acc);
    }
    int c = tn * 16 + lo;
#pragma unroll
    for (int r = 0; r < 8; ++r) {
      int p = tm * 16 + hi * 8 + r;
      if (p < P) {
        int i = p / PW, j = p - i * PW;
        size_t g = xbase + (size_t)c * (Hh * Ww) + i * Ww + j;
        float xv = x[g];            // L2-hot: this block just streamed it
        float ec = acc[r];
        ecg [g] = ec;
        outp[g] = xv * (beta * ec + xv);
      }
    }
  }
}

extern "C" void kernel_launch(void* const* d_in, const int* in_sizes, int n_in,
                              void* d_out, int out_size, void* d_ws, size_t ws_size,
                              hipStream_t stream) {
  (void)in_sizes; (void)n_in; (void)out_size; (void)d_ws; (void)ws_size;
  const float* x    = (const float*)d_in[0];
  const float* beta = (const float*)d_in[1];
  // d_in[2], d_in[3] are ph, pw (compile-time 7 here)

  float* outp = (float*)d_out;
  const size_t nImg = (size_t)Bb * Cc * Hh * Ww;     // 51,380,224
  const size_t nSq  = (size_t)Bb * Mm * P * P;       //  9,834,496
  float* scg  = outp + nImg;
  float* covg = scg  + nSq;
  float* lg   = covg + nSq;
  float* ecg  = lg   + nSq;

  dim3 grid(Bb * Mm), block(NT);
  psam_kernel<<<grid, block, 0, stream>>>(x, beta, outp, scg, covg, lg, ecg);
}
